// SpatialSumLayer_16088947491591
// MI455X (gfx1250) — compile-verified, matrix operand order checked by reading, not run on hardware
//
#include <hip/hip_runtime.h>

// y[b,o,hw] = log( sum_c exp(x[b,c,hw]-xmax[b,hw]) * exp(W[o,c]-wmx[o]) )
//             + xmax[b,hw] - lse[o]
// GEMM M=o(256) x N=hw x K=c(256) in f16 WMMA 16x16x32, f32 accumulate.

typedef __attribute__((ext_vector_type(16))) _Float16    v16h;
typedef __attribute__((ext_vector_type(8)))  float       v8f;
typedef __attribute__((ext_vector_type(4)))  unsigned int v4u;

#define C_DIM    256     // input channels (K)
#define O_DIM    256     // output channels (GEMM "M")
#define HW       4096    // H*W
#define M_TILE   64      // hw positions per block (GEMM "N" tile)
#define AS_STRIDE 264    // f16 LDS row stride (16B-aligned, bank-spread)

// Load one 16x(K=32-chunk) f16 fragment: two contiguous 16B chunks per lane,
// at K offsets {ks .. ks+7} and {ks+16 .. ks+23} within the 32-wide K chunk.
__device__ __forceinline__ v16h ld_frag(const _Float16* p) {
  union { v16h h; v4u u[2]; } r;
  r.u[0] = *(const v4u*)(p);
  r.u[1] = *(const v4u*)(p + 16);
  return r.h;
}

// ---------------- kernel 1: weight prep (O blocks x C threads) -------------
__global__ void __launch_bounds__(C_DIM)
prep_weights(const float* __restrict__ weight, _Float16* __restrict__ ewf,
             float* __restrict__ wmax) {
  __shared__ float red[C_DIM];
  const int o = blockIdx.x;
  const int t = threadIdx.x;
  const float wv = weight[(size_t)o * C_DIM + t];

  red[t] = wv;
  __syncthreads();
  #pragma unroll
  for (int s = C_DIM / 2; s > 0; s >>= 1) {
    if (t < s) red[t] = fmaxf(red[t], red[t + s]);
    __syncthreads();
  }
  const float mx = red[0];
  __syncthreads();

  const float e = __expf(wv - mx);
  red[t] = e;
  __syncthreads();
  #pragma unroll
  for (int s = C_DIM / 2; s > 0; s >>= 1) {
    if (t < s) red[t] = red[t] + red[t + s];
    __syncthreads();
  }
  ewf[(size_t)o * C_DIM + t] = (_Float16)e;
  if (t == 0) wmax[o] = -__logf(red[0]);   // = -lse[o]
}

// ---------------- kernel 2: main fused LSE-GEMM ----------------------------
__global__ void __launch_bounds__(256)
lse_mix_kernel(const float* __restrict__ x, const _Float16* __restrict__ ewf,
               const float* __restrict__ wmax, float* __restrict__ out) {
  __shared__ _Float16 sA[M_TILE * AS_STRIDE];   // exp(x - xmax), [m][k] f16
  __shared__ float    sXmax[M_TILE];
  __shared__ float    sWmax[O_DIM];
  __shared__ float    sPart[16][M_TILE];

  const int tid  = threadIdx.x;
  const int bIdx = blockIdx.x >> 6;          // HW/M_TILE = 64 tiles per batch
  const int hw0  = (blockIdx.x & 63) * M_TILE;
  const float* xb = x + (size_t)bIdx * C_DIM * HW + hw0;

  sWmax[tid] = wmax[tid];                    // 256 threads == O_DIM

  const int kg = tid >> 4;                   // 0..15 : k-group
  const int mq = (tid & 15) * 4;             // 0,4,...,60 : 4 m's per thread

  // ---- pass 1: per-position max over all 256 channels (streams from HBM)
  float r0 = -INFINITY, r1 = -INFINITY, r2 = -INFINITY, r3 = -INFINITY;
  #pragma unroll
  for (int i = 0; i < 16; ++i) {
    const int k = i * 16 + kg;
    const float4 v = *(const float4*)(xb + (size_t)k * HW + mq);
    r0 = fmaxf(r0, v.x); r1 = fmaxf(r1, v.y);
    r2 = fmaxf(r2, v.z); r3 = fmaxf(r3, v.w);
  }
  sPart[kg][mq + 0] = r0; sPart[kg][mq + 1] = r1;
  sPart[kg][mq + 2] = r2; sPart[kg][mq + 3] = r3;
  __syncthreads();
  if (tid < M_TILE) {
    float m = sPart[0][tid];
    #pragma unroll
    for (int g = 1; g < 16; ++g) m = fmaxf(m, sPart[g][tid]);
    sXmax[tid] = m;
  }
  __syncthreads();

  // ---- pass 2: exp(x - xmax) -> f16 LDS tile (re-read hits L2)
  const float xm0 = sXmax[mq + 0], xm1 = sXmax[mq + 1];
  const float xm2 = sXmax[mq + 2], xm3 = sXmax[mq + 3];
  #pragma unroll
  for (int i = 0; i < 16; ++i) {
    const int k = i * 16 + kg;
    const float4 v = *(const float4*)(xb + (size_t)k * HW + mq);
    sA[(mq + 0) * AS_STRIDE + k] = (_Float16)__expf(v.x - xm0);
    sA[(mq + 1) * AS_STRIDE + k] = (_Float16)__expf(v.y - xm1);
    sA[(mq + 2) * AS_STRIDE + k] = (_Float16)__expf(v.z - xm2);
    sA[(mq + 3) * AS_STRIDE + k] = (_Float16)__expf(v.w - xm3);
  }
  __syncthreads();

  // ---- WMMA: wave w owns o in [w*32, w*32+32) x all 64 hw positions
  const int w    = tid >> 5;
  const int lane = tid & 31;
  const int l16  = lane & 15;
  const int ks   = (lane < 16) ? 0 : 8;      // K sub-offset per ISA layout

  v8f acc[2][4];
  #pragma unroll
  for (int i = 0; i < 2; ++i)
    #pragma unroll
    for (int j = 0; j < 4; ++j) acc[i][j] = (v8f)0.0f;

  #pragma unroll
  for (int kk = 0; kk < 8; ++kk) {           // K chunks of 32
    const int kc = kk * 32;
    v16h aw[2];                              // A: ew rows (M = o)
    #pragma unroll
    for (int i = 0; i < 2; ++i) {
      const int o = w * 32 + i * 16 + l16;
      aw[i] = ld_frag(ewf + (size_t)o * C_DIM + kc + ks);
    }
    v16h bx[4];                              // B: exp-x cols (N = hw)
    #pragma unroll
    for (int j = 0; j < 4; ++j) {
      const int m = j * 16 + l16;
      bx[j] = ld_frag(&sA[m * AS_STRIDE + kc + ks]);
    }
    #pragma unroll
    for (int i = 0; i < 2; ++i)
      #pragma unroll
      for (int j = 0; j < 4; ++j)
        acc[i][j] = __builtin_amdgcn_wmma_f32_16x16x32_f16(
            false, aw[i], false, bx[j], (short)0, acc[i][j], false, false);
  }

  // ---- epilogue: log(sum) + xmax[m] + wmax[o]; lane-contiguous NT stores
  const int oHalf = (lane < 16) ? 0 : 8;
  #pragma unroll
  for (int i = 0; i < 2; ++i) {
    const int oBase = w * 32 + i * 16 + oHalf;
    #pragma unroll
    for (int j = 0; j < 4; ++j) {
      const int m  = j * 16 + l16;
      const float xm = sXmax[m];
      #pragma unroll
      for (int r = 0; r < 8; ++r) {
        const int oo = oBase + r;
        const float val = __logf(acc[i][j][r]) + xm + sWmax[oo];
        float* p = out + ((size_t)bIdx * O_DIM + oo) * HW + hw0 + m;
        __builtin_nontemporal_store(val, p);
      }
    }
  }
}

// ---------------- host launcher --------------------------------------------
extern "C" void kernel_launch(void* const* d_in, const int* in_sizes, int n_in,
                              void* d_out, int out_size, void* d_ws, size_t ws_size,
                              hipStream_t stream) {
  const float* x      = (const float*)d_in[0];   // (B, C, H, W) f32
  const float* weight = (const float*)d_in[1];   // (O, C, 1, 1) f32
  float* out = (float*)d_out;                    // (B, O, H, W) f32

  _Float16* ewf  = (_Float16*)d_ws;                                   // O*C f16
  float*    wmx  = (float*)((char*)d_ws + (size_t)O_DIM * C_DIM * 2); // O f32

  const int Bn = in_sizes[0] / (C_DIM * HW);     // 32

  prep_weights<<<O_DIM, C_DIM, 0, stream>>>(weight, ewf, wmx);
  const int nblocks = Bn * (HW / M_TILE);        // 2048
  lse_mix_kernel<<<nblocks, 256, 0, stream>>>(x, ewf, wmx, out);
}
